// Local_Receptive_Field_33148557590653
// MI455X (gfx1250) — matrix operand order repaired and checked by reading, not compiled, not run on hardware
//
#include <hip/hip_runtime.h>

typedef _Float16 h8  __attribute__((ext_vector_type(8)));
typedef _Float16 h16 __attribute__((ext_vector_type(16)));
typedef float    f2  __attribute__((ext_vector_type(2)));
typedef float    f4  __attribute__((ext_vector_type(4)));
typedef float    f8  __attribute__((ext_vector_type(8)));

#define BATCH   512
#define NNODES  1024
#define MTOT    (BATCH * NNODES)   // 524288 rows
#define HIDDEN  90
#define HP      96                 // padded hidden (6 tiles of 16)
#define NLAY    4

// ---- LDS layout (bytes) ----
#define OFF_WTIN   0                              // [96][32] f16 : W_in^T              (6144 B)
#define OFF_WT     6144                           // [4][96][96] f16 : W_h^T           (73728 B)
#define OFF_WTOUT  79872                          // [16][96] f16 : W_out^T (row0 real) (3072 B)
#define OFF_BIN    82944                          // [96] f32                            (384 B)
#define OFF_BH     83328                          // [4][96] f32                        (1536 B)
#define OFF_ACT    84864                          // 8 waves x 2 tiles x [16][96] f16  (49152 B)
#define SMEM_BYTES (84864 + 8 * 2 * 16 * HP * 2)  // 134016 B

// A fragment (16x32 f16): lane l -> row m=l&15; 16 contiguous halves at p.
__device__ __forceinline__ h16 lds_load_a(const _Float16* p) {
  const h8* v = (const h8*)p;
  h8 lo = v[0];
  h8 hi = v[1];
  return __builtin_shufflevector(lo, hi, 0,1,2,3,4,5,6,7,8,9,10,11,12,13,14,15);
}

// B fragment (32x16 f16): 16 contiguous halves at p (lane = column, lane-half = K group).
__device__ __forceinline__ h16 lds_load_b(const _Float16* p) {
  const h8* v = (const h8*)p;
  h8 lo = v[0];
  h8 hi = v[1];
  return __builtin_shufflevector(lo, hi, 0,1,2,3,4,5,6,7,8,9,10,11,12,13,14,15);
}

__device__ __forceinline__ f8 splat8(float v) {
  f8 r;
#pragma unroll
  for (int i = 0; i < 8; ++i) r[i] = v;
  return r;
}

// Single-instruction ReLU: v_med3_f32(x, 0, +inf) clamps to [0, inf).
__device__ __forceinline__ float relu1(float x) {
  return __builtin_amdgcn_fmed3f(x, 0.f, __builtin_inff());
}

// ReLU then packed RTZ f32->f16 convert; rows m and m+1 (stride 96 halves).
__device__ __forceinline__ void relu_store_pair(_Float16* dst, float a, float b) {
  auto p = __builtin_amdgcn_cvt_pkrtz(relu1(a), relu1(b));  // __fp16 x2
  dst[0]  = (_Float16)p[0];
  dst[HP] = (_Float16)p[1];
}

// Store one 16x16 D tile into act (f16, ReLU'd). base = act + (lh*8)*HP + column.
__device__ __forceinline__ void store_tile(_Float16* base, const f8& acc) {
#pragma unroll
  for (int q = 0; q < 4; ++q)
    relu_store_pair(base + (2 * q) * HP, acc[2 * q], acc[2 * q + 1]);
}

__device__ __forceinline__ h16 build_input_frag(const float* __restrict__ x,
                                                const float* __restrict__ ctrl,
                                                int row) {
  // lane-half 0 covers electrodes 0..15 (7 real); lane-half 1 covers 16..31 (all zero)
  h16 a = {};
  const int b  = row >> 10;
  const int n  = row & 1023;
  const int pi = n >> 5, pj = n & 31;
  const f2* top = (const f2*)(x + b * 4096 + pi * 128 + pj * 2);
  const f2* bot = (const f2*)(x + b * 4096 + pi * 128 + 64 + pj * 2);
  const float* cp = ctrl + n * 3;
  f2 t = top[0], bo = bot[0];
  a[0] = (_Float16)cp[0];
  a[1] = (_Float16)t[0];
  a[2] = (_Float16)t[1];
  a[3] = (_Float16)bo[0];
  a[4] = (_Float16)bo[1];
  a[5] = (_Float16)cp[1];
  a[6] = (_Float16)cp[2];
  return a;
}

__global__ __launch_bounds__(256) void dnpu_lrf_mlp_kernel(
    const float* __restrict__ x,     // [512,1,64,64]
    const float* __restrict__ ctrl,  // [1024,3]
    const float* __restrict__ Win,   // [7,90]
    const float* __restrict__ bin,   // [90]
    const float* __restrict__ Wh,    // [4,90,90]
    const float* __restrict__ bh,    // [4,90]
    const float* __restrict__ Wout,  // [90,1]
    const float* __restrict__ bout,  // [1]
    float* __restrict__ out)         // [524288]
{
  extern __shared__ char smem[];
  _Float16* sWtIn  = (_Float16*)(smem + OFF_WTIN);
  _Float16* sWt    = (_Float16*)(smem + OFF_WT);
  _Float16* sWtOut = (_Float16*)(smem + OFF_WTOUT);
  float*    sBin   = (float*)(smem + OFF_BIN);
  float*    sBh    = (float*)(smem + OFF_BH);
  _Float16* sAct   = (_Float16*)(smem + OFF_ACT);

  const int tid = threadIdx.x;

  // ---- stage weights (transposed, f16, zero-padded) into LDS ----
  for (int idx = tid; idx < HP * 32; idx += 256) {          // W_in^T [n][e]
    int n = idx >> 5, e = idx & 31;
    float v = (e < 7 && n < HIDDEN) ? Win[e * HIDDEN + n] : 0.f;
    sWtIn[idx] = (_Float16)v;
  }
  for (int idx = tid; idx < NLAY * HP * HP; idx += 256) {   // W_h^T [i][n][k]
    int i = idx / (HP * HP);
    int r = idx - i * HP * HP;
    int n = r / HP;
    int k = r - n * HP;
    float v = (n < HIDDEN && k < HIDDEN) ? Wh[i * HIDDEN * HIDDEN + k * HIDDEN + n] : 0.f;
    sWt[idx] = (_Float16)v;
  }
  for (int idx = tid; idx < 16 * HP; idx += 256) {          // W_out^T [n][k], n==0 real
    int n = idx / HP, k = idx - n * HP;
    float v = (n == 0 && k < HIDDEN) ? Wout[k] : 0.f;
    sWtOut[idx] = (_Float16)v;
  }
  for (int idx = tid; idx < HP; idx += 256)
    sBin[idx] = (idx < HIDDEN) ? bin[idx] : 0.f;
  for (int idx = tid; idx < NLAY * HP; idx += 256) {
    int i = idx / HP, k = idx - i * HP;
    sBh[idx] = (k < HIDDEN) ? bh[i * HIDDEN + k] : 0.f;
  }
  __syncthreads();

  const int lane = tid & 31;
  const int wv   = tid >> 5;
  const int nl   = lane & 15;     // N column / M row within tile
  const int lh   = lane >> 4;     // lane-half selector
  _Float16* act0 = sAct + wv * (2 * 16 * HP);
  _Float16* act1 = act0 + 16 * HP;
  const float bo = bout[0];

  // per-lane LDS bases
  _Float16* aBase0 = act0 + nl * HP + lh * 16;          // A-frag loads
  _Float16* aBase1 = act1 + nl * HP + lh * 16;
  _Float16* sBase0 = act0 + (lh * 8) * HP;              // D-tile stores (+ column)
  _Float16* sBase1 = act1 + (lh * 8) * HP;

  for (int p = 0; p < 2; ++p) {
    const int R0 = ((blockIdx.x * 8 + wv) * 4 + p * 2) * 16;  // two 16-row blocks
    const int R1 = R0 + 16;

    // ---- input layer: two A frags share every B frag ----
    h16 a0 = {}, a1 = {};
    if (lh == 0) {
      a0 = build_input_frag(x, ctrl, R0 + nl);
      a1 = build_input_frag(x, ctrl, R1 + nl);
    }

#pragma unroll
    for (int t = 0; t < 6; ++t) {
      f8 cb = splat8(sBin[t * 16 + nl]);                  // bias folded into C operand
      h16 bf = lds_load_b(sWtIn + (t * 16 + nl) * 32 + lh * 16);
      f8 d0 = __builtin_amdgcn_wmma_f32_16x16x32_f16(false, a0, false, bf,
                                                     (short)0, cb, false, false);
      f8 d1 = __builtin_amdgcn_wmma_f32_16x16x32_f16(false, a1, false, bf,
                                                     (short)0, cb, false, false);
      store_tile(sBase0 + t * 16 + nl, d0);
      store_tile(sBase1 + t * 16 + nl, d1);
    }

    // ---- 4 hidden layers: 2 x (16x96 @ 96x96), shared B frags ----
    for (int L = 0; L < NLAY; ++L) {
      const _Float16* w = sWt + L * HP * HP;
      h16 af0[3], af1[3];
#pragma unroll
      for (int c = 0; c < 3; ++c) {                       // load ALL A frags before overwrite
        af0[c] = lds_load_a(aBase0 + c * 32);
        af1[c] = lds_load_a(aBase1 + c * 32);
      }
#pragma unroll
      for (int t = 0; t < 6; ++t) {
        f8 cb = splat8(sBh[L * HP + t * 16 + nl]);
        f8 d0 = cb, d1 = cb;
#pragma unroll
        for (int c = 0; c < 3; ++c) {
          h16 bf = lds_load_b(w + (t * 16 + nl) * HP + c * 32 + lh * 16);
          d0 = __builtin_amdgcn_wmma_f32_16x16x32_f16(false, af0[c], false, bf,
                                                      (short)0, d0, false, false);
          d1 = __builtin_amdgcn_wmma_f32_16x16x32_f16(false, af1[c], false, bf,
                                                      (short)0, d1, false, false);
        }
        store_tile(sBase0 + t * 16 + nl, d0);
        store_tile(sBase1 + t * 16 + nl, d1);
      }
    }

    // ---- output layer: 2 x (16x96 @ 96x16), only column 0 real; b_out in C ----
    {
      h16 af0[3], af1[3];
#pragma unroll
      for (int c = 0; c < 3; ++c) {
        af0[c] = lds_load_a(aBase0 + c * 32);
        af1[c] = lds_load_a(aBase1 + c * 32);
      }
      f8 d0 = splat8(bo), d1 = splat8(bo);
#pragma unroll
      for (int c = 0; c < 3; ++c) {
        h16 bf = lds_load_b(sWtOut + nl * HP + c * 32 + lh * 16);
        d0 = __builtin_amdgcn_wmma_f32_16x16x32_f16(false, af0[c], false, bf,
                                                    (short)0, d0, false, false);
        d1 = __builtin_amdgcn_wmma_f32_16x16x32_f16(false, af1[c], false, bf,
                                                    (short)0, d1, false, false);
      }
      if (nl == 0) {                 // lanes 0/16 hold column n=0, rows m = r + 8*lh
        f4 lo0 = {d0[0], d0[1], d0[2], d0[3]};
        f4 hi0 = {d0[4], d0[5], d0[6], d0[7]};
        f4 lo1 = {d1[0], d1[1], d1[2], d1[3]};
        f4 hi1 = {d1[4], d1[5], d1[6], d1[7]};
        *(f4*)(out + R0 + lh * 8)     = lo0;
        *(f4*)(out + R0 + lh * 8 + 4) = hi0;
        *(f4*)(out + R1 + lh * 8)     = lo1;
        *(f4*)(out + R1 + lh * 8 + 4) = hi1;
      }
    }
  }
}

extern "C" void kernel_launch(void* const* d_in, const int* in_sizes, int n_in,
                              void* d_out, int out_size, void* d_ws, size_t ws_size,
                              hipStream_t stream) {
  const float* x    = (const float*)d_in[0];
  const float* ctrl = (const float*)d_in[1];
  const float* Win  = (const float*)d_in[2];
  const float* bin  = (const float*)d_in[3];
  const float* Wh   = (const float*)d_in[4];
  const float* bh   = (const float*)d_in[5];
  const float* Wout = (const float*)d_in[6];
  const float* bout = (const float*)d_in[7];
  // d_in[8]=data_idx, d_in[9]=ctrl_idx are constant placements, hardcoded above.
  float* out = (float*)d_out;

  (void)in_sizes; (void)n_in; (void)out_size; (void)d_ws; (void)ws_size;

  (void)hipFuncSetAttribute(reinterpret_cast<const void*>(dnpu_lrf_mlp_kernel),
                            hipFuncAttributeMaxDynamicSharedMemorySize, SMEM_BYTES);

  // 32768 M-blocks of 16 rows; 8 waves/WG, 4 blocks/wave (2 concurrent x 2 iter) -> 1024 WGs
  dim3 grid(MTOT / (16 * 8 * 4));
  dim3 block(256);
  dnpu_lrf_mlp_kernel<<<grid, block, SMEM_BYTES, stream>>>(
      x, ctrl, Win, bin, Wh, bh, Wout, bout, out);
}
